// RuleDecoder_74526272520873
// MI455X (gfx1250) — compile-verified, matrix operand order checked by reading, not run on hardware
//
#include <hip/hip_runtime.h>
#include <hip/hip_bf16.h>

typedef __attribute__((ext_vector_type(16))) __bf16 v16bf;
typedef __attribute__((ext_vector_type(8)))  __bf16 v8bf;
typedef __attribute__((ext_vector_type(8)))  float  v8f;
typedef __attribute__((ext_vector_type(4)))  float  v4f;
typedef __attribute__((ext_vector_type(4)))  int    v4i;
typedef __attribute__((address_space(3))) v4i lds_v4i;

#define D_DIM 512
#define H_DIM 1024
#define NRELS 2048
#define BR    32768          // 16 * 2048 rows
#define NWAVES 8

// padded LDS strides (elements)
#define XF_STRIDE 520        // f32
#define AT_STRIDE 520        // f32
#define HB_STRIDE 520        // bf16
#define HR_STRIDE 1040       // bf16

#if defined(__gfx1250__) && \
    __has_builtin(__builtin_amdgcn_global_load_async_to_lds_b128) && \
    __has_builtin(__builtin_amdgcn_s_wait_asynccnt)
#define HAVE_ASYNC_LDS 1
#else
#define HAVE_ASYNC_LDS 0
#endif

// ---------------- fragment loaders (CDNA5 16-bit WMMA layouts) ----------------
// A (16x32 bf16): lane l -> row (l&15); first 8 vals K=kb+((l>>4)<<3)..+7, second 8 at +16.
__device__ __forceinline__ v16bf a_frag_f32(const float* base, int stride, int kb, int lane) {
    int row = lane & 15;
    int k0  = kb + ((lane >> 4) << 3);
    const float* p = base + row * stride + k0;
    v4f a0 = *(const v4f*)(p + 0);
    v4f a1 = *(const v4f*)(p + 4);
    v4f b0 = *(const v4f*)(p + 16);
    v4f b1 = *(const v4f*)(p + 20);
    v16bf r;
#pragma unroll
    for (int i = 0; i < 4; ++i) {
        r[i]      = (__bf16)a0[i];
        r[4 + i]  = (__bf16)a1[i];
        r[8 + i]  = (__bf16)b0[i];
        r[12 + i] = (__bf16)b1[i];
    }
    return r;
}

__device__ __forceinline__ v16bf a_frag_bf16(const __bf16* base, int stride, int kb, int lane) {
    int row = lane & 15;
    int k0  = kb + ((lane >> 4) << 3);
    const __bf16* p = base + row * stride + k0;
    v8bf lo = *(const v8bf*)p;
    v8bf hi = *(const v8bf*)(p + 16);
    v16bf r;
#pragma unroll
    for (int i = 0; i < 8; ++i) { r[i] = lo[i]; r[8 + i] = hi[i]; }
    return r;
}

// B (32x16 bf16) for logits = h @ Wᵀ with W row-major [N][K]: column n of B = row (nb+n) of W.
__device__ __forceinline__ v16bf b_frag(const __bf16* w, int ldk, int nbrow, int kb, int lane) {
    int col = lane & 15;
    int k0  = kb + ((lane >> 4) << 3);
    const __bf16* p = w + (size_t)(nbrow + col) * ldk + k0;
    v8bf lo = *(const v8bf*)p;
    v8bf hi = *(const v8bf*)(p + 16);
    v16bf r;
#pragma unroll
    for (int i = 0; i < 8; ++i) { r[i] = lo[i]; r[8 + i] = hi[i]; }
    return r;
}

#define WMMA_BF16(acc, a, b) \
    __builtin_amdgcn_wmma_f32_16x16x32_bf16(false, (a), false, (b), (short)0, (acc), false, false)

// ---------------- prep kernels ----------------
// Wc[n][k] = sum_j o_w[n][j] * v_w[j][k]   (bf16 output)
__global__ void fuse_weights_kernel(const float* __restrict__ o_w,
                                    const float* __restrict__ v_w,
                                    __bf16* __restrict__ Wc) {
    int idx = blockIdx.x * 256 + threadIdx.x;
    int n = idx >> 9, k = idx & 511;
    float s = 0.f;
    for (int j = 0; j < D_DIM; ++j) s += o_w[n * D_DIM + j] * v_w[j * D_DIM + k];
    Wc[idx] = (__bf16)s;
}

// bc[n] = o_b[n] + sum_j o_w[n][j] * v_b[j]
__global__ void fuse_bias_kernel(const float* __restrict__ o_w,
                                 const float* __restrict__ v_b,
                                 const float* __restrict__ o_b,
                                 float* __restrict__ bc) {
    int n = blockIdx.x * 256 + threadIdx.x;
    if (n < D_DIM) {
        float s = o_b[n];
        for (int j = 0; j < D_DIM; ++j) s += o_w[n * D_DIM + j] * v_b[j];
        bc[n] = s;
    }
}

__global__ void cvt_weights_kernel(const float* __restrict__ w1,
                                   const float* __restrict__ w2,
                                   __bf16* __restrict__ w1b,
                                   __bf16* __restrict__ w2b) {
    int idx = blockIdx.x * 256 + threadIdx.x;
    const int N1 = H_DIM * D_DIM;         // 524288
    if (idx < N1) w1b[idx] = (__bf16)w1[idx];
    else          w2b[idx - N1] = (__bf16)w2[idx - N1];
}

// ---------------- main fused kernel ----------------
__global__ __launch_bounds__(256) void rule_decoder_kernel(
    const float* __restrict__ x,
    const float* __restrict__ na_emb,
    const float* __restrict__ ln_g, const float* __restrict__ ln_b,
    const float* __restrict__ b1,   const float* __restrict__ b2,
    const __bf16* __restrict__ Wc,  const float* __restrict__ bc,
    const __bf16* __restrict__ w1b, const __bf16* __restrict__ w2b,
    float* __restrict__ out)
{
    __shared__ float  xf[16 * XF_STRIDE];        // x tile (f32)             32.5 KB
    __shared__ float  at[16 * AT_STRIDE];        // attn tile (f32)          32.5 KB
    __shared__ __bf16 hb[16 * HB_STRIDE];        // LN output (bf16)         16.3 KB
    __shared__ __bf16 Hr[2][16 * HR_STRIDE];     // relu hidden, both steps  65   KB
    __shared__ float  wmaxS[2][NWAVES][16];
    __shared__ float  wsumS[2][NWAVES][16];
    __shared__ int    widxS[2][NWAVES][16];
    __shared__ float  mres[2][16];
    __shared__ int    ires[2][16];

    const int tid  = threadIdx.x;
    const int lane = tid & 31;
    const int wave = tid >> 5;
    const size_t row0 = (size_t)blockIdx.x * 16;

    // ---- stage x tile: 16 rows x 512 f32 = 2048 x 16B across 256 threads ----
#if HAVE_ASYNC_LDS
    for (int i = tid; i < 16 * 128; i += 256) {
        int r  = i >> 7;
        int c4 = i & 127;
        __builtin_amdgcn_global_load_async_to_lds_b128(
            (v4i*)(x + (row0 + r) * D_DIM + c4 * 4),
            (lds_v4i*)(&xf[r * XF_STRIDE + c4 * 4]),
            /*offset=*/0, /*cpol=*/0);
    }
    __builtin_amdgcn_s_wait_asynccnt(0);
#else
    for (int i = tid; i < 16 * 128; i += 256) {
        int r  = i >> 7;
        int c4 = i & 127;
        v4f v = *(const v4f*)(x + (row0 + r) * D_DIM + c4 * 4);
        *(v4f*)(&xf[r * XF_STRIDE + c4 * 4]) = v;
    }
#endif
    __syncthreads();

    // ---- attn = x @ Wcᵀ + bc  (512 WMMAs/block) ----
    for (int nt = wave; nt < D_DIM / 16; nt += NWAVES) {
        v8f acc = {};
#pragma unroll 4
        for (int kt = 0; kt < D_DIM / 32; ++kt) {
            v16bf a = a_frag_f32(xf, XF_STRIDE, kt * 32, lane);
            v16bf b = b_frag(Wc, D_DIM, nt * 16, kt * 32, lane);
            acc = WMMA_BF16(acc, a, b);
        }
        int col   = nt * 16 + (lane & 15);
        float bia = bc[col];
        int rbase = (lane >> 4) << 3;
#pragma unroll
        for (int j = 0; j < 8; ++j)
            at[(rbase + j) * AT_STRIDE + col] = acc[j] + bia;
    }
    __syncthreads();

    // ---- build relu-hidden for BOTH steps (so GEMM2 streams w2 only once) ----
    for (int step = 0; step < 2; ++step) {
        // LayerNorm(attn + state): each wave owns 2 rows, 32 lanes x 16 elems
        for (int rr = 0; rr < 2; ++rr) {
            int r = wave * 2 + rr;
            float s = 0.f, ss = 0.f;
            float vals[16];
#pragma unroll
            for (int i = 0; i < 16; ++i) {
                int c = lane + i * 32;
                float st = (step == 0) ? na_emb[c] : xf[r * XF_STRIDE + c];
                float v  = at[r * AT_STRIDE + c] + st;
                vals[i] = v; s += v; ss += v * v;
            }
#pragma unroll
            for (int off = 16; off >= 1; off >>= 1) {
                s  += __shfl_xor(s,  off, 32);
                ss += __shfl_xor(ss, off, 32);
            }
            float mean = s * (1.f / 512.f);
            float var  = ss * (1.f / 512.f) - mean * mean;
            float rstd = __frsqrt_rn(var + 1e-5f);
#pragma unroll
            for (int i = 0; i < 16; ++i) {
                int c = lane + i * 32;
                hb[r * HB_STRIDE + c] = (__bf16)((vals[i] - mean) * rstd * ln_g[c] + ln_b[c]);
            }
        }
        __syncthreads();

        // GEMM1: Hr[step] = relu(hb @ w1ᵀ + b1)   (1024 WMMAs/block)
        for (int nt = wave; nt < H_DIM / 16; nt += NWAVES) {
            v8f acc = {};
#pragma unroll 4
            for (int kt = 0; kt < D_DIM / 32; ++kt) {
                v16bf a = a_frag_bf16(hb, HB_STRIDE, kt * 32, lane);
                v16bf b = b_frag(w1b, D_DIM, nt * 16, kt * 32, lane);
                acc = WMMA_BF16(acc, a, b);
            }
            int col   = nt * 16 + (lane & 15);
            float bia = b1[col];
            int rbase = (lane >> 4) << 3;
#pragma unroll
            for (int j = 0; j < 8; ++j) {
                float v = acc[j] + bia;
                Hr[step][(rbase + j) * HR_STRIDE + col] = (__bf16)(v > 0.f ? v : 0.f);
            }
        }
        __syncthreads();   // also protects hb before next step's LN overwrite
    }

    // ---- combined GEMM2 for both steps: each w2 B-fragment feeds 2 WMMAs ----
    // (8192 WMMAs/block; online softmax-max/argmax per step)
    float rM[2][8], rS[2][8]; int rI[2][8];
#pragma unroll
    for (int s = 0; s < 2; ++s)
#pragma unroll
        for (int j = 0; j < 8; ++j) { rM[s][j] = -1e30f; rS[s][j] = 0.f; rI[s][j] = 0; }

    for (int nt = wave; nt < NRELS / 16; nt += NWAVES) {
        v8f acc0 = {}, acc1 = {};
#pragma unroll 4
        for (int kt = 0; kt < H_DIM / 32; ++kt) {
            v16bf b  = b_frag(w2b, H_DIM, nt * 16, kt * 32, lane);   // loaded ONCE
            v16bf a0 = a_frag_bf16(Hr[0], HR_STRIDE, kt * 32, lane);
            v16bf a1 = a_frag_bf16(Hr[1], HR_STRIDE, kt * 32, lane);
            acc0 = WMMA_BF16(acc0, a0, b);
            acc1 = WMMA_BF16(acc1, a1, b);
        }
        int col = nt * 16 + (lane & 15);
        float bia = b2[col];
#pragma unroll
        for (int s = 0; s < 2; ++s) {
#pragma unroll
            for (int j = 0; j < 8; ++j) {
                float l = (s == 0 ? acc0[j] : acc1[j]) + bia;
                // tile row-max + argmax across the 16 lanes holding this row
                float tv = l; int ti = col;
#pragma unroll
                for (int off = 8; off >= 1; off >>= 1) {
                    float ov = __shfl_xor(tv, off, 16);
                    int   oi = __shfl_xor(ti, off, 16);
                    if (ov > tv || (ov == tv && oi < ti)) { tv = ov; ti = oi; }
                }
                float nM = fmaxf(rM[s][j], tv);
                float e  = __expf(l - nM);
#pragma unroll
                for (int off = 8; off >= 1; off >>= 1) e += __shfl_xor(e, off, 16);
                rS[s][j] = rS[s][j] * __expf(rM[s][j] - nM) + e;
                if (tv > rM[s][j]) rI[s][j] = ti;
                rM[s][j] = nM;
            }
        }
    }
    if ((lane & 15) == 0) {
        int rbase = (lane >> 4) << 3;
#pragma unroll
        for (int s = 0; s < 2; ++s)
#pragma unroll
            for (int j = 0; j < 8; ++j) {
                wmaxS[s][wave][rbase + j] = rM[s][j];
                wsumS[s][wave][rbase + j] = rS[s][j];
                widxS[s][wave][rbase + j] = rI[s][j];
            }
    }
    __syncthreads();

    // cross-wave merge: one thread per row, per step
    if (tid < 32) {
        int s = tid >> 4;
        int r = tid & 15;
        float M = -1e30f; int I = 0;
        for (int w = 0; w < NWAVES; ++w) {
            float m = wmaxS[s][w][r];
            int   i = widxS[s][w][r];
            if (m > M) { M = m; I = i; }
            else if (m == M && i < I) I = i;
        }
        float S = 0.f;
        for (int w = 0; w < NWAVES; ++w)
            S += wsumS[s][w][r] * __expf(wmaxS[s][w][r] - M);
        mres[s][r] = 1.f / S;   // softmax max prob = exp(M-M)/S
        ires[s][r] = I;
    }
    __syncthreads();

    // ---- early-stop scoring + output ----
    if (tid < 16) {
        float m0 = mres[0][tid]; int i0 = ires[0][tid];
        float m1 = mres[1][tid]; int i1 = ires[1][tid];
        bool c0 = (i0 != 0) && (m0 >= 0.1f);
        bool c1 = c0 && (i1 != 0) && (m1 >= 0.1f);
        float score = c0 ? (c1 ? m0 * m1 : m0) : 0.f;
        int len = (int)c0 + (int)c1;
        size_t g = row0 + tid;
        out[g]            = score;
        out[BR + g]       = (float)i0;
        out[2 * BR + g]   = (float)i1;
        out[3 * BR + g]   = (float)len;
    }
}

// ---------------- launcher ----------------
extern "C" void kernel_launch(void* const* d_in, const int* in_sizes, int n_in,
                              void* d_out, int out_size, void* d_ws, size_t ws_size,
                              hipStream_t stream) {
    const float* x    = (const float*)d_in[0];   // [16,2048,512]
    const float* na   = (const float*)d_in[1];   // [512]
    const float* v_w  = (const float*)d_in[2];   // [512,512]
    const float* v_b  = (const float*)d_in[3];   // [512]
    const float* o_w  = (const float*)d_in[4];   // [512,512]
    const float* o_b  = (const float*)d_in[5];   // [512]
    const float* lng  = (const float*)d_in[6];   // [512]
    const float* lnb  = (const float*)d_in[7];   // [512]
    const float* w1   = (const float*)d_in[8];   // [1024,512]
    const float* b1   = (const float*)d_in[9];   // [1024]
    const float* w2   = (const float*)d_in[10];  // [2048,1024]
    const float* b2   = (const float*)d_in[11];  // [2048]

    char* ws = (char*)d_ws;
    __bf16* Wc  = (__bf16*)ws;                                   // 512*512 bf16
    __bf16* w1b = (__bf16*)(ws + (size_t)512 * 512 * 2);         // 1024*512 bf16
    __bf16* w2b = (__bf16*)(ws + (size_t)512 * 512 * 2
                               + (size_t)1024 * 512 * 2);        // 2048*1024 bf16
    float*  bc  = (float*)(ws + (size_t)512 * 512 * 2
                              + (size_t)1024 * 512 * 2
                              + (size_t)2048 * 1024 * 2);        // 512 f32

    fuse_weights_kernel<<<(512 * 512) / 256, 256, 0, stream>>>(o_w, v_w, Wc);
    fuse_bias_kernel<<<2, 256, 0, stream>>>(o_w, v_b, o_b, bc);
    cvt_weights_kernel<<<(1024 * 512 + 2048 * 1024) / 256, 256, 0, stream>>>(w1, w2, w1b, w2b);

    rule_decoder_kernel<<<BR / 16, 256, 0, stream>>>(
        x, na, lng, lnb, b1, b2, Wc, bc, w1b, w2b, (float*)d_out);
}
